// Attention_83313775608434
// MI455X (gfx1250) — matrix operand order repaired and checked by reading, not compiled, not run on hardware
//
#include <hip/hip_runtime.h>
#include <math.h>

// ---------------------------------------------------------------------------
// Types for CDNA5 WMMA (wave32)
// ---------------------------------------------------------------------------
typedef __attribute__((ext_vector_type(16))) __bf16 v16bf;
typedef __attribute__((ext_vector_type(8)))  __bf16 v8bf;
typedef __attribute__((ext_vector_type(8)))  float  v8f;
typedef __attribute__((ext_vector_type(4)))  float  v4f;

#define WMMA_BF16(a, b, c) \
    __builtin_amdgcn_wmma_f32_16x16x32_bf16(false, (a), false, (b), (short)0, (c), false, false)

// ---------------------------------------------------------------------------
// Kernel 1: C[M,N] = A[M,K] @ W[N,K]^T   (fp32 in, bf16 WMMA, fp32 out)
// Workgroup: 256 threads = 8 waves; block tile 128(M) x 128(N); K-step 32.
// Wave grid 4(M) x 2(N): each wave computes 32x64 -> 2 A-frags x 4 B-frags,
// 8 WMMAs per K-step. Global loads for step k+1 are register-staged while
// step k's WMMAs execute (single LDS buffer, latency hidden in registers).
// ---------------------------------------------------------------------------
__global__ __launch_bounds__(256)
void gemm_xwT(const float* __restrict__ A, const float* __restrict__ W,
              float* __restrict__ C, int M, int N, int K)
{
    __shared__ __bf16 As[128][40];   // 128 x 32 (+pad)
    __shared__ __bf16 Bs[128][40];   // 128 x 32 (+pad)

    const int tid  = threadIdx.x;
    const int wave = tid >> 5;
    const int lane = tid & 31;
    const int half = lane >> 4;
    const int l16  = lane & 15;
    const int mg   = (wave & 3) * 32;   // wave's M offset in tile
    const int ng   = (wave >> 2) * 64;  // wave's N offset in tile
    const int rowBase = blockIdx.y * 128;
    const int colBase = blockIdx.x * 128;

    // staging-slot geometry: 256 threads x 4 float4 covers 128x32 floats
    const int sr = tid >> 3;            // row 0..31 (+ i*32)
    const int sc = (tid & 7) << 2;      // col 0..28 step 4

    v8f acc[2][4] = {};
    v4f ra[4], rb[4];

    // prologue: stage K-step 0 into registers
    #pragma unroll
    for (int i = 0; i < 4; ++i) {
        ra[i] = *(const v4f*)&A[(size_t)(rowBase + sr + i * 32) * K + sc];
        rb[i] = *(const v4f*)&W[(size_t)(colBase + sr + i * 32) * K + sc];
    }

    for (int k0 = 0; k0 < K; k0 += 32) {
        // ---- registers -> LDS (fp32 -> bf16) ----
        #pragma unroll
        for (int i = 0; i < 4; ++i) {
            int r = sr + i * 32;
            As[r][sc + 0] = (__bf16)ra[i][0];
            As[r][sc + 1] = (__bf16)ra[i][1];
            As[r][sc + 2] = (__bf16)ra[i][2];
            As[r][sc + 3] = (__bf16)ra[i][3];
            Bs[r][sc + 0] = (__bf16)rb[i][0];
            Bs[r][sc + 1] = (__bf16)rb[i][1];
            Bs[r][sc + 2] = (__bf16)rb[i][2];
            Bs[r][sc + 3] = (__bf16)rb[i][3];
        }
        __syncthreads();

        // ---- issue global loads for K-step k0+32 (wait lands after WMMAs) ----
        if (k0 + 32 < K) {
            #pragma unroll
            for (int i = 0; i < 4; ++i) {
                ra[i] = *(const v4f*)&A[(size_t)(rowBase + sr + i * 32) * K + k0 + 32 + sc];
                rb[i] = *(const v4f*)&W[(size_t)(colBase + sr + i * 32) * K + k0 + 32 + sc];
            }
        }
        // distance-2 prefetch into GL2 (global_prefetch_b8)
        if (k0 + 64 < K) {
            __builtin_prefetch(&A[(size_t)(rowBase + sr) * K + k0 + 64 + sc], 0, 1);
            __builtin_prefetch(&W[(size_t)(colBase + sr) * K + k0 + 64 + sc], 0, 1);
        }

        // ---- A fragments (16x32, contiguous k = 16*half + s), 2 per wave ----
        v16bf af[2];
        #pragma unroll
        for (int f = 0; f < 2; ++f) {
            int m = mg + f * 16 + l16;
            v8bf lo = *(const v8bf*)&As[m][half * 16];
            v8bf hi = *(const v8bf*)&As[m][half * 16 + 8];
            #pragma unroll
            for (int i = 0; i < 8; ++i) { af[f][i] = lo[i]; af[f][i + 8] = hi[i]; }
        }
        // ---- 8 WMMAs per K-step ----
        #pragma unroll
        for (int t = 0; t < 4; ++t) {
            v16bf bfrag = *(const v16bf*)&Bs[ng + t * 16 + l16][half * 16];
            #pragma unroll
            for (int f = 0; f < 2; ++f)
                acc[f][t] = WMMA_BF16(af[f], bfrag, acc[f][t]);
        }
        __syncthreads();
    }

    // ---- epilogue: hoisted base address, walk rows with adds ----
    #pragma unroll
    for (int f = 0; f < 2; ++f) {
        size_t base = (size_t)(rowBase + mg + f * 16 + half * 8) * N
                      + colBase + ng + l16;
        #pragma unroll
        for (int r = 0; r < 8; ++r) {
            #pragma unroll
            for (int t = 0; t < 4; ++t)
                C[base + t * 16] = acc[f][t][r];
            base += N;
        }
    }
}

// ---------------------------------------------------------------------------
// Kernel 2: RoPE applied in place to Q and K.
// ---------------------------------------------------------------------------
__global__ __launch_bounds__(256)
void rope_kernel(float* __restrict__ q, float* __restrict__ k,
                 const float* __restrict__ fcos, const float* __restrict__ fsin,
                 int S, int D, int HD, size_t npairs)
{
    size_t i = (size_t)blockIdx.x * blockDim.x + threadIdx.x;
    if (i >= npairs) return;
    int pairsPerRow = D >> 1;
    size_t row = i / pairsPerRow;
    int pr = (int)(i - row * pairsPerRow);
    int j  = pr & ((HD >> 1) - 1);
    int s  = (int)(row % S);
    float c  = fcos[s * (HD >> 1) + j];
    float sn = fsin[s * (HD >> 1) + j];
    size_t base = row * (size_t)D + 2 * pr;
    float q0 = q[base], q1 = q[base + 1];
    q[base]     = q0 * c - q1 * sn;
    q[base + 1] = q0 * sn + q1 * c;
    float k0 = k[base], k1 = k[base + 1];
    k[base]     = k0 * c - k1 * sn;
    k[base + 1] = k0 * sn + k1 * c;
}

// ---------------------------------------------------------------------------
// Kernel 3: causal flash attention, one (b,h) per blockIdx.y, 64 q-rows/block.
// 128 threads = 4 waves; wave w owns q rows [qt*64 + w*16, +16).
// K/V tiles for block j+1 register-staged while block j computes.
// ---------------------------------------------------------------------------
__global__ __launch_bounds__(128)
void flash_attn(const float* __restrict__ Q, const float* __restrict__ Km,
                const float* __restrict__ Vm, float* __restrict__ O,
                int B, int S, int H, int HD, float scale)
{
    __shared__ __bf16 Ks[32][72];      // [key][hd]
    __shared__ __bf16 Vt[64][40];      // [hd][key] (transposed)
    __shared__ __bf16 Ps[4][16][40];   // per-wave P scratch

    const int tid  = threadIdx.x;
    const int wave = tid >> 5;
    const int lane = tid & 31;
    const int half = lane >> 4;
    const int l16  = lane & 15;

    const int bh = blockIdx.y;
    const int b  = bh / H;
    const int h  = bh - b * H;
    const int qt = blockIdx.x;
    const int qbase = qt * 64 + wave * 16;
    const int D = H * HD;
    const size_t headOff = (size_t)h * HD;

    // staging-slot geometry: 128 threads x 4 float4 covers 32x64 floats
    const int skr = tid >> 4;           // key 0..7 (+ i*8)
    const int sc4 = (tid & 15) << 2;    // hd 0..60 step 4

    // ---- Q fragments: 2 K-steps over HD=64, persistent in registers ----
    v16bf qf[2];
    {
        int m = qbase + l16;
        const float* qrow = Q + ((size_t)b * S + m) * D + headOff;
        #pragma unroll
        for (int ks = 0; ks < 2; ++ks) {
            v4f lo0 = *(const v4f*)&qrow[ks * 32 + half * 16];
            v4f lo1 = *(const v4f*)&qrow[ks * 32 + half * 16 + 4];
            v4f hi0 = *(const v4f*)&qrow[ks * 32 + half * 16 + 8];
            v4f hi1 = *(const v4f*)&qrow[ks * 32 + half * 16 + 12];
            #pragma unroll
            for (int i = 0; i < 4; ++i) {
                qf[ks][i]      = (__bf16)lo0[i];
                qf[ks][4 + i]  = (__bf16)lo1[i];
                qf[ks][8 + i]  = (__bf16)hi0[i];
                qf[ks][12 + i] = (__bf16)hi1[i];
            }
        }
    }

    v8f   o[4] = {};
    float m_i[8], l_i[8];
    #pragma unroll
    for (int r = 0; r < 8; ++r) { m_i[r] = -3.0e38f; l_i[r] = 0.0f; }

    const int jend = qt * 2 + 2;          // key blocks of 32, causal bound
    v4f kreg[4], vreg[4];

    // prologue: stage key-block 0
    #pragma unroll
    for (int i = 0; i < 4; ++i) {
        size_t gro = ((size_t)b * S + skr + i * 8) * D + headOff + sc4;
        kreg[i] = *(const v4f*)&Km[gro];
        vreg[i] = *(const v4f*)&Vm[gro];
    }

    for (int j = 0; j < jend; ++j) {
        const int kbase = j * 32;

        // ---- registers -> LDS: K row-major, V transposed ----
        #pragma unroll
        for (int i = 0; i < 4; ++i) {
            int kr = skr + i * 8;
            Ks[kr][sc4 + 0] = (__bf16)kreg[i][0];
            Ks[kr][sc4 + 1] = (__bf16)kreg[i][1];
            Ks[kr][sc4 + 2] = (__bf16)kreg[i][2];
            Ks[kr][sc4 + 3] = (__bf16)kreg[i][3];
            Vt[sc4 + 0][kr] = (__bf16)vreg[i][0];
            Vt[sc4 + 1][kr] = (__bf16)vreg[i][1];
            Vt[sc4 + 2][kr] = (__bf16)vreg[i][2];
            Vt[sc4 + 3][kr] = (__bf16)vreg[i][3];
        }
        __syncthreads();

        // ---- issue global loads for key-block j+1 ----
        if (j + 1 < jend) {
            #pragma unroll
            for (int i = 0; i < 4; ++i) {
                size_t gro = ((size_t)b * S + kbase + 32 + skr + i * 8) * D + headOff + sc4;
                kreg[i] = *(const v4f*)&Km[gro];
                vreg[i] = *(const v4f*)&Vm[gro];
            }
        }

        // ---- scores: S = Q(16x64) @ K^T(64x32) -> two 16x16 tiles ----
        v8f st[2] = {};
        #pragma unroll
        for (int t = 0; t < 2; ++t) {
            #pragma unroll
            for (int ks = 0; ks < 2; ++ks) {
                v16bf kb = *(const v16bf*)&Ks[t * 16 + l16][ks * 32 + half * 16];
                st[t] = WMMA_BF16(qf[ks], kb, st[t]);
            }
        }

        // ---- scale + causal mask + online softmax update ----
        float sv[2][8];
        #pragma unroll
        for (int t = 0; t < 2; ++t)
            #pragma unroll
            for (int r = 0; r < 8; ++r) {
                int sq  = qbase + half * 8 + r;
                int key = kbase + t * 16 + l16;
                sv[t][r] = (key <= sq) ? st[t][r] * scale : -1.0e30f;
            }
        #pragma unroll
        for (int r = 0; r < 8; ++r) {
            float mx = fmaxf(sv[0][r], sv[1][r]);
            #pragma unroll
            for (int d = 1; d < 16; d <<= 1)
                mx = fmaxf(mx, __shfl_xor(mx, d, 32));
            float mn = fmaxf(m_i[r], mx);
            float p0 = __expf(sv[0][r] - mn);
            float p1 = __expf(sv[1][r] - mn);
            float ps = p0 + p1;
            #pragma unroll
            for (int d = 1; d < 16; d <<= 1)
                ps += __shfl_xor(ps, d, 32);
            float alpha = __expf(m_i[r] - mn);
            l_i[r] = l_i[r] * alpha + ps;
            m_i[r] = mn;
            #pragma unroll
            for (int t4 = 0; t4 < 4; ++t4) o[t4][r] *= alpha;
            Ps[wave][half * 8 + r][l16]      = (__bf16)p0;
            Ps[wave][half * 8 + r][16 + l16] = (__bf16)p1;
        }

        // ---- P (C-layout in LDS) -> A-fragment, then O += P @ V ----
        v16bf pf;
        {
            v8bf lo = *(const v8bf*)&Ps[wave][l16][half * 16];
            v8bf hi = *(const v8bf*)&Ps[wave][l16][half * 16 + 8];
            #pragma unroll
            for (int i = 0; i < 8; ++i) { pf[i] = lo[i]; pf[i + 8] = hi[i]; }
        }
        #pragma unroll
        for (int t4 = 0; t4 < 4; ++t4) {
            v16bf vb = *(const v16bf*)&Vt[t4 * 16 + l16][half * 16];
            o[t4] = WMMA_BF16(pf, vb, o[t4]);
        }
        __syncthreads();
    }

    // ---- epilogue: divide by row sums, store fp32 ----
    #pragma unroll
    for (int r = 0; r < 8; ++r) {
        float inv = 1.0f / l_i[r];
        int sq = qbase + half * 8 + r;
        size_t rowo = ((size_t)b * S + sq) * D + headOff;
        #pragma unroll
        for (int t4 = 0; t4 < 4; ++t4)
            O[rowo + t4 * 16 + l16] = o[t4][r] * inv;
    }
}

// ---------------------------------------------------------------------------
// Host-side orchestration
// ---------------------------------------------------------------------------
extern "C" void kernel_launch(void* const* d_in, const int* in_sizes, int n_in,
                              void* d_out, int out_size, void* d_ws, size_t ws_size,
                              hipStream_t stream)
{
    const float* x    = (const float*)d_in[0];
    const float* wq   = (const float*)d_in[1];
    const float* wk   = (const float*)d_in[2];
    const float* wv   = (const float*)d_in[3];
    const float* wo   = (const float*)d_in[4];
    const float* fcos = (const float*)d_in[5];
    const float* fsin = (const float*)d_in[6];
    float* out = (float*)d_out;

    const int Bv = 2, Sv = 2048, Dv = 2048, Hv = 32, HDv = 64;
    const int Mrows = Bv * Sv;                 // 4096
    const size_t tsz = (size_t)Mrows * Dv;     // elements per [B,S,D] tensor

    float* Qb = (float*)d_ws;                  // needs 4 * tsz * 4 bytes = 128 MiB
    float* Kb = Qb + tsz;
    float* Vb = Kb + tsz;
    float* Ob = Vb + tsz;

    dim3 gg(Dv / 128, Mrows / 128);            // 16 x 32 workgroups per GEMM

    gemm_xwT<<<gg, 256, 0, stream>>>(x, wq, Qb, Mrows, Dv, Dv);
    gemm_xwT<<<gg, 256, 0, stream>>>(x, wk, Kb, Mrows, Dv, Dv);
    gemm_xwT<<<gg, 256, 0, stream>>>(x, wv, Vb, Mrows, Dv, Dv);

    size_t npairs = tsz / 2;
    rope_kernel<<<(unsigned)((npairs + 255) / 256), 256, 0, stream>>>(
        Qb, Kb, fcos, fsin, Sv, Dv, HDv, npairs);

    float scale = (float)((log(2048.0) / log(1024.0)) / 8.0);   // log_scale / sqrt(HD)
    flash_attn<<<dim3(Sv / 64, Bv * Hv), 128, 0, stream>>>(
        Qb, Kb, Vb, Ob, Bv, Sv, Hv, HDv, scale);

    gemm_xwT<<<gg, 256, 0, stream>>>(Ob, wo, out, Mrows, Dv, Dv);
}
